// DeepKoopmanSplit_48112223650188
// MI455X (gfx1250) — compile-verified, matrix operand order checked by reading, not run on hardware
//
#include <hip/hip_runtime.h>
#include <hip/hip_bf16.h>

// ---------------------------------------------------------------------------
// DeepKoopmanSplit on MI455X (gfx1250): bf16 WMMA, weight-stationary LDS MLPs,
// TDM (tensor_load_to_lds) weight staging, row-chunked persistent blocks,
// dual-accumulator WMMA interleaving to hide WMMA->WMMA hazards.
// ---------------------------------------------------------------------------

typedef __attribute__((ext_vector_type(16))) __bf16 v16bf;
typedef __attribute__((ext_vector_type(8)))  float  v8f;
typedef unsigned int u32x4 __attribute__((ext_vector_type(4)));
typedef int          i32x4 __attribute__((ext_vector_type(4)));
typedef int          i32x8 __attribute__((ext_vector_type(8)));

#define TPB   256
#define WAVES 8
#define RPB   64   // rows per 64-row chunk

#define BATCH 8192LL
#define MSTEP 64
#define BM    (BATCH * MSTEP)          // 524288 rows

// d_out flat offsets (floats)
#define O_XK 0LL
#define O_XT 98304LL
#define O_ZP 6389760LL
#define O_XP 73498624LL
#define O_ZT 79790080LL

#if defined(__HIP_DEVICE_COMPILE__) && \
    __has_builtin(__builtin_amdgcn_tensor_load_to_lds) && \
    __has_builtin(__builtin_amdgcn_s_wait_tensorcnt)
#define HAS_TDM 1
#else
#define HAS_TDM 0
#endif

// ---------------------------------------------------------------------------
// TDM: DMA a linear bf16 buffer (multiple of 256 elems) global -> LDS as a
// 2-D tile: rowlen=256 elems (512B), rows=total/256. D# per ISA 08 §8.
// ---------------------------------------------------------------------------
#if HAS_TDM
__device__ __forceinline__ void tdm_load_linear(const __bf16* gsrc, __bf16* ldst,
                                                int total_elems) {
  const int rowlen = 256;
  const int rows = total_elems >> 8;
  unsigned long long ga = (unsigned long long)(uintptr_t)gsrc;
  unsigned lds = (unsigned)(uintptr_t)ldst;
  u32x4 g0;
  g0[0] = 1u;                                    // count=1, user desc
  g0[1] = lds;                                   // lds_addr (bytes)
  g0[2] = (unsigned)(ga & 0xFFFFFFFFu);          // global_addr[31:0]
  g0[3] = (unsigned)((ga >> 32) & 0x01FFFFFFu) | (2u << 30);  // [56:32] | type=2
  i32x8 g1;
  g1[0] = 0x00010000;                            // wg_mask=0, data_size=1 (2B)
  g1[1] = (rowlen & 0xFFFF) << 16;               // tensor_dim0 lo16 (bits 63:48)
  g1[2] = ((rowlen >> 16) & 0xFFFF) | ((rows & 0xFFFF) << 16);   // dim0 hi, dim1 lo
  g1[3] = ((rows >> 16) & 0xFFFF) | ((rowlen & 0xFFFF) << 16);   // dim1 hi, tile_dim0
  g1[4] = rows & 0xFFFF;                         // tile_dim1 (tile_dim2=0)
  g1[5] = rowlen;                                // tensor_dim0_stride lo32
  g1[6] = 0;
  g1[7] = 0;
  i32x4 z4 = {0, 0, 0, 0};
#if defined(__clang_major__) && __clang_major__ >= 23
  i32x8 z8 = {0, 0, 0, 0, 0, 0, 0, 0};
  __builtin_amdgcn_tensor_load_to_lds(g0, g1, z4, z4, z8, 0);
#else
  __builtin_amdgcn_tensor_load_to_lds(g0, g1, z4, z4, 0);
#endif
}
#endif

__device__ __forceinline__ void stage_lds(const __bf16* gsrc, __bf16* ldst,
                                          int total_elems, int tid, int wave) {
#if HAS_TDM
  if (wave == 0) tdm_load_linear(gsrc, ldst, total_elems);
  (void)tid;
#else
  const uint4* s = (const uint4*)gsrc;
  uint4* d = (uint4*)ldst;
  for (int i = tid; i < (total_elems >> 3); i += TPB) d[i] = s[i];
#endif
}

__device__ __forceinline__ void stage_wait(int wave) {
#if HAS_TDM
  if (wave == 0) __builtin_amdgcn_s_wait_tensorcnt(0);
#else
  (void)wave;
#endif
}

// ---------------------------------------------------------------------------
// fragment helpers (CDNA5 WMMA 16x16x32 bf16 layouts, wave32)
// ---------------------------------------------------------------------------
__device__ __forceinline__ v16bf lds_load16(const __bf16* p) {
  union { v16bf v; uint4 q[2]; } u;
  u.q[0] = *(const uint4*)(p);
  u.q[1] = *(const uint4*)(p + 8);
  return u.v;
}

// A (16x32) fragment from row-major bf16 tile in LDS.
__device__ __forceinline__ v16bf load_a_frag(const __bf16* tile, int ld, int lane) {
  v16bf a;
  const int m  = lane & 15;
  const int hi = (lane >> 4) & 1;
  const __bf16* r = tile + m * ld;
#pragma unroll
  for (int p = 0; p < 8; ++p) {
    int k0 = (p < 4) ? (2 * p + 8 * hi) : (16 + 2 * (p - 4) + 8 * hi);
    a[2 * p]     = r[k0];
    a[2 * p + 1] = r[k0 + 1];
  }
  return a;
}

__device__ __forceinline__ v8f wmma_bf16(v16bf a, v16bf b, v8f c) {
  return __builtin_amdgcn_wmma_f32_16x16x32_bf16(false, a, false, b, (short)0, c,
                                                 false, false);
}

// ---------------------------------------------------------------------------
// dense hidden layer: [64 x 32*KS] x [32*KS x 256] -> sOut (bias+relu), LDS.
// Each wave owns row-tile rt=wave&3 and 8 n-tiles; processes n-tiles in pairs
// with two independent accumulators to fill WMMA hazard slots.
// ---------------------------------------------------------------------------
template <int KS>
__device__ __forceinline__ void dense_layer_lds(const __bf16* sIn, int ldin,
                                                const __bf16* sW, const float* sB,
                                                __bf16* sOut, int wave, int lane) {
  const int rt = wave & 3;
  const int ntbase = (wave >> 2) * 8;
  v16bf af[KS];
#pragma unroll
  for (int ks = 0; ks < KS; ++ks)
    af[ks] = load_a_frag(sIn + (rt * 16) * ldin + ks * 32, ldin, lane);
#pragma unroll
  for (int jj = 0; jj < 4; ++jj) {
    const int nt0 = ntbase + 2 * jj, nt1 = nt0 + 1;
    v8f a0 = {}, a1 = {};
#pragma unroll
    for (int ks = 0; ks < KS; ++ks) {
      v16bf b0 = lds_load16(sW + ((((ks * 16 + nt0) * 32) + lane) << 4));
      v16bf b1 = lds_load16(sW + ((((ks * 16 + nt1) * 32) + lane) << 4));
      a0 = wmma_bf16(af[ks], b0, a0);
      a1 = wmma_bf16(af[ks], b1, a1);
    }
    const int n = lane & 15, hi = lane >> 4;
#pragma unroll
    for (int r = 0; r < 8; ++r) {
      int m = r + 8 * hi;
      float x0 = fmaxf(a0[r] + sB[nt0 * 16 + n], 0.f);
      float x1 = fmaxf(a1[r] + sB[nt1 * 16 + n], 0.f);
      sOut[(rt * 16 + m) * 256 + nt0 * 16 + n] = (__bf16)x0;
      sOut[(rt * 16 + m) * 256 + nt1 * 16 + n] = (__bf16)x1;
    }
  }
}

// ---------------------------------------------------------------------------
// prepass: fp32 weight W[N][K] -> bf16, transposed+padded, WMMA-B fragment
// order: tile (ks,nt) stores 32 lanes x 16 values contiguously.
// ---------------------------------------------------------------------------
__global__ void conv_wt_frag(const float* __restrict__ w, __bf16* __restrict__ dst,
                             int N, int K, int Npad, int Kpad) {
  int i = blockIdx.x * TPB + threadIdx.x;
  int total = Npad * Kpad;
  if (i >= total) return;
  int v    = i & 15;
  int lane = (i >> 4) & 31;
  int tile = i >> 9;
  int ntiles = Npad >> 4;
  int ks = tile / ntiles;
  int nt = tile - ks * ntiles;
  int nl = lane & 15, hi = (lane >> 4) & 1;
  int p = v >> 1, odd = v & 1;
  int k0 = (p < 4) ? (2 * p + 8 * hi) : (16 + 2 * (p - 4) + 8 * hi);
  int k = ks * 32 + k0 + odd;
  int n = nt * 16 + nl;
  float val = (n < N && k < K) ? w[(long long)n * K + k] : 0.f;
  dst[i] = (__bf16)val;
}

// ---------------------------------------------------------------------------
// fused 3-layer MLP (Linear->ReLU->Linear->ReLU->Linear), weight-stationary.
// rchunk x 64 rows per block, 8 waves; hidden width fixed at 256.
// ---------------------------------------------------------------------------
__global__ void __launch_bounds__(TPB) mlp3_kernel(
    const float* __restrict__ in, long long in_rs, int in_co, int K_in,
    long long n_rows,
    const __bf16* __restrict__ w0f, int K0pad,
    const __bf16* __restrict__ w1f,
    const __bf16* __restrict__ w2f, int N2pad, int N2act,
    const float* __restrict__ b0g, const float* __restrict__ b1g,
    const float* __restrict__ b2g,
    float* __restrict__ out, long long out_rs, int out_co, int rchunk) {
  extern __shared__ char smem[];
  const int tid  = threadIdx.x;
  const int lane = tid & 31;
  const int wave = tid >> 5;

  __bf16* sW0 = (__bf16*)smem;                 // K0pad*256
  __bf16* sW1 = sW0 + K0pad * 256;             // 65536
  __bf16* sW2 = sW1 + 65536;                   // 256*N2pad
  __bf16* sX  = sW2 + 256 * N2pad;             // 64*K0pad
  __bf16* sH0 = sX + RPB * K0pad;              // 64*256
  __bf16* sH1 = sH0 + RPB * 256;               // 64*256
  float*  sB0 = (float*)(sH1 + RPB * 256);
  float*  sB1 = sB0 + 256;
  float*  sB2 = sB1 + 256;

  // stage weights once per block (TDM on gfx1250; fallback uint4 copies)
  stage_lds(w0f, sW0, K0pad * 256, tid, wave);
  stage_lds(w1f, sW1, 65536, tid, wave);
  stage_lds(w2f, sW2, 256 * N2pad, tid, wave);
  for (int i = tid; i < 256; i += TPB) { sB0[i] = b0g[i]; sB1[i] = b1g[i]; }
  for (int i = tid; i < N2pad; i += TPB) sB2[i] = (i < N2act) ? b2g[i] : 0.f;
  stage_wait(wave);

  for (int c = 0; c < rchunk; ++c) {
    const long long row0 = ((long long)blockIdx.x * rchunk + c) * RPB;
    if (row0 >= n_rows) break;

    // stage input rows, f32 -> bf16, zero-padded
    for (int i = tid; i < RPB * K0pad; i += TPB) {
      int r = i / K0pad, k = i - r * K0pad;
      long long row = row0 + r;
      float v = 0.f;
      if (row < n_rows && k < K_in) v = in[row * in_rs + in_co + k];
      sX[i] = (__bf16)v;
    }
    // prefetch next chunk's input rows into cache
    {
      long long nrow0 = row0 + RPB;
      if (nrow0 < n_rows) {
        for (int i = tid * 32; i < RPB * (int)in_rs; i += TPB * 32)
          __builtin_prefetch(in + nrow0 * in_rs + i, 0, 1);
      }
    }
    __syncthreads();

    // ---- layer 0: [64 x K0pad] x [K0pad x 256] -> sH0 (bias+relu) ----
    if (K0pad == 32)
      dense_layer_lds<1>(sX, K0pad, sW0, sB0, sH0, wave, lane);
    else
      dense_layer_lds<3>(sX, K0pad, sW0, sB0, sH0, wave, lane);
    __syncthreads();

    // ---- layer 1: [64x256] x [256x256] -> sH1 (bias+relu) ----
    dense_layer_lds<8>(sH0, 256, sW1, sB1, sH1, wave, lane);
    __syncthreads();

    // ---- layer 2: [64x256] x [256xN2pad] -> global (bias, no relu) ----
    {
      const int nt2 = N2pad >> 4;
      const int rt = wave & 3;
      const int n = lane & 15, hi = lane >> 4;
      v16bf af[8];
#pragma unroll
      for (int ks = 0; ks < 8; ++ks)
        af[ks] = load_a_frag(sH1 + (rt * 16) * 256 + ks * 32, 256, lane);
      int nt = wave >> 2;
      // paired n-tiles (this wave's tiles are nt, nt+2, nt+4, ...)
      for (; nt + 2 < nt2; nt += 4) {
        const int nt0 = nt, nt1 = nt + 2;
        v8f a0 = {}, a1 = {};
#pragma unroll
        for (int ks = 0; ks < 8; ++ks) {
          v16bf b0 = lds_load16(sW2 + ((((ks * nt2 + nt0) * 32) + lane) << 4));
          v16bf b1 = lds_load16(sW2 + ((((ks * nt2 + nt1) * 32) + lane) << 4));
          a0 = wmma_bf16(af[ks], b0, a0);
          a1 = wmma_bf16(af[ks], b1, a1);
        }
#pragma unroll
        for (int r = 0; r < 8; ++r) {
          int m = r + 8 * hi;
          long long row = row0 + rt * 16 + m;
          int c0 = nt0 * 16 + n, c1 = nt1 * 16 + n;
          if (row < n_rows) {
            if (c0 < N2act) out[row * out_rs + out_co + c0] = a0[r] + sB2[c0];
            if (c1 < N2act) out[row * out_rs + out_co + c1] = a1[r] + sB2[c1];
          }
        }
      }
      // single tail tile
      if (nt < nt2) {
        v8f a0 = {};
#pragma unroll
        for (int ks = 0; ks < 8; ++ks)
          a0 = wmma_bf16(af[ks],
                         lds_load16(sW2 + ((((ks * nt2 + nt) * 32) + lane) << 4)),
                         a0);
        int col = nt * 16 + n;
#pragma unroll
        for (int r = 0; r < 8; ++r) {
          int m = r + 8 * hi;
          long long row = row0 + rt * 16 + m;
          if (row < n_rows && col < N2act)
            out[row * out_rs + out_co + col] = a0[r] + sB2[col];
        }
      }
    }
    __syncthreads();
  }
}

// ---------------------------------------------------------------------------
// Koopman recurrence: z_{i+1} = z_i A^T + u_i B^T, 64 rows/block, M steps.
// A^T (128x128) and padded B^T (32x128) stationary in LDS, z double-buffered.
// ---------------------------------------------------------------------------
__global__ void __launch_bounds__(TPB) koopman_kernel(
    const float* __restrict__ zk, const float* __restrict__ u,
    const __bf16* __restrict__ atf, const __bf16* __restrict__ btf,
    float* __restrict__ zpred, int M) {
  extern __shared__ char smem[];
  const int tid = threadIdx.x, lane = tid & 31, wave = tid >> 5;
  const long long row0 = (long long)blockIdx.x * RPB;

  __bf16* sAt = (__bf16*)smem;   // 128*128 frag-ordered
  __bf16* sBt = sAt + 16384;     // 32*128 frag-ordered
  __bf16* sZ0 = sBt + 4096;      // 64*128 row-major
  __bf16* sZ1 = sZ0 + 8192;
  __bf16* sU  = sZ1 + 8192;      // 64*32 row-major (K padded 4->32)

  stage_lds(atf, sAt, 16384, tid, wave);
  stage_lds(btf, sBt, 4096, tid, wave);
  for (int i = tid; i < RPB * 128; i += TPB) {
    int r = i >> 7, c = i & 127;
    sZ0[i] = (__bf16)zk[(row0 + r) * 128 + c];
  }
  stage_wait(wave);

  for (int step = 0; step < M; ++step) {
    for (int i = tid; i < RPB * 32; i += TPB) {
      int r = i >> 5, k = i & 31;
      float v = (k < 4) ? u[((row0 + r) * M + step) * 4 + k] : 0.f;
      sU[i] = (__bf16)v;
    }
    __syncthreads();
    __bf16* zc = (step & 1) ? sZ1 : sZ0;
    __bf16* zn = (step & 1) ? sZ0 : sZ1;

    const int rt = wave & 3;
    const int ntbase = (wave >> 2) * 4;
    v16bf az[4];
#pragma unroll
    for (int ks = 0; ks < 4; ++ks)
      az[ks] = load_a_frag(zc + (rt * 16) * 128 + ks * 32, 128, lane);
    v16bf au = load_a_frag(sU + (rt * 16) * 32, 32, lane);

#pragma unroll
    for (int jj = 0; jj < 2; ++jj) {
      const int nt0 = ntbase + 2 * jj, nt1 = nt0 + 1;
      v8f a0 = {}, a1 = {};
#pragma unroll
      for (int ks = 0; ks < 4; ++ks) {
        v16bf b0 = lds_load16(sAt + ((((ks * 8 + nt0) * 32) + lane) << 4));
        v16bf b1 = lds_load16(sAt + ((((ks * 8 + nt1) * 32) + lane) << 4));
        a0 = wmma_bf16(az[ks], b0, a0);
        a1 = wmma_bf16(az[ks], b1, a1);
      }
      a0 = wmma_bf16(au, lds_load16(sBt + (((nt0 * 32) + lane) << 4)), a0);
      a1 = wmma_bf16(au, lds_load16(sBt + (((nt1 * 32) + lane) << 4)), a1);
      const int n = lane & 15, hi = lane >> 4;
      const int c0 = nt0 * 16 + n, c1 = nt1 * 16 + n;
#pragma unroll
      for (int r = 0; r < 8; ++r) {
        int m = r + 8 * hi;
        long long row = row0 + rt * 16 + m;
        zpred[(row * M + step) * 128 + c0] = a0[r];
        zpred[(row * M + step) * 128 + c1] = a1[r];
        zn[(rt * 16 + m) * 128 + c0] = (__bf16)a0[r];
        zn[(rt * 16 + m) * 128 + c1] = (__bf16)a1[r];
      }
    }
    __syncthreads();
  }
}

// ---------------------------------------------------------------------------
extern "C" void kernel_launch(void* const* d_in, const int* in_sizes, int n_in,
                              void* d_out, int out_size, void* d_ws, size_t ws_size,
                              hipStream_t stream) {
  (void)in_sizes; (void)n_in; (void)out_size; (void)ws_size;
  const float* x_k    = (const float*)d_in[0];
  const float* u_seq  = (const float*)d_in[1];
  const float* x_next = (const float*)d_in[2];
  float* outp = (float*)d_out;

  __bf16* wsb = (__bf16*)d_ws;
  size_t off = 0;
  auto conv = [&](const float* w, int N, int K, int Npad, int Kpad) -> __bf16* {
    __bf16* dst = wsb + off;
    int total = Npad * Kpad;
    off += (size_t)total;
    conv_wt_frag<<<(total + TPB - 1) / TPB, TPB, 0, stream>>>(w, dst, N, K, Npad, Kpad);
    return dst;
  };

  // encoder_pos: 3->256->256->32
  __bf16* ep0 = conv((const float*)d_in[3], 256, 3, 256, 32);
  __bf16* ep1 = conv((const float*)d_in[5], 256, 256, 256, 256);
  __bf16* ep2 = conv((const float*)d_in[7], 32, 256, 32, 256);
  // encoder_rest: 9->256->256->96
  __bf16* er0 = conv((const float*)d_in[9], 256, 9, 256, 32);
  __bf16* er1 = conv((const float*)d_in[11], 256, 256, 256, 256);
  __bf16* er2 = conv((const float*)d_in[13], 96, 256, 96, 256);
  // decoder_pos: 32->256->256->3
  __bf16* dp0 = conv((const float*)d_in[15], 256, 32, 256, 32);
  __bf16* dp1 = conv((const float*)d_in[17], 256, 256, 256, 256);
  __bf16* dp2 = conv((const float*)d_in[19], 3, 256, 16, 256);
  // decoder_rest: 96->256->256->9
  __bf16* dr0 = conv((const float*)d_in[21], 256, 96, 256, 96);
  __bf16* dr1 = conv((const float*)d_in[23], 256, 256, 256, 256);
  __bf16* dr2 = conv((const float*)d_in[25], 9, 256, 16, 256);
  // Koopman operators
  __bf16* atf = conv((const float*)d_in[27], 128, 128, 128, 128);
  __bf16* btf = conv((const float*)d_in[28], 128, 4, 128, 32);

  // z_k scratch (f32 [B,128]) after weights; 16B aligned
  float* zk = (float*)(wsb + ((off + 7) & ~(size_t)7));

  auto mlp = [&](const float* in, long long irs, int ico, int kin, long long nrows,
                 const __bf16* w0f, int k0pad, const __bf16* w1f, const __bf16* w2f,
                 int n2pad, int n2act, const float* b0, const float* b1,
                 const float* b2, float* out, long long ors, int oco, int rchunk) {
    size_t smem = (size_t)k0pad * 256 * 2 + 65536 * 2 + 256 * (size_t)n2pad * 2 +
                  RPB * (size_t)k0pad * 2 + 2 * RPB * 256 * 2 +
                  256 * 4 + 256 * 4 + (size_t)n2pad * 4;
    long long rows_per_blk = (long long)RPB * rchunk;
    dim3 grid((unsigned)((nrows + rows_per_blk - 1) / rows_per_blk));
    mlp3_kernel<<<grid, TPB, smem, stream>>>(in, irs, ico, kin, nrows, w0f, k0pad,
                                             w1f, w2f, n2pad, n2act, b0, b1, b2,
                                             out, ors, oco, rchunk);
  };

  const float* ep_b0 = (const float*)d_in[4],  *ep_b1 = (const float*)d_in[6],
             * ep_b2 = (const float*)d_in[8];
  const float* er_b0 = (const float*)d_in[10], *er_b1 = (const float*)d_in[12],
             * er_b2 = (const float*)d_in[14];
  const float* dp_b0 = (const float*)d_in[16], *dp_b1 = (const float*)d_in[18],
             * dp_b2 = (const float*)d_in[20];
  const float* dr_b0 = (const float*)d_in[22], *dr_b1 = (const float*)d_in[24],
             * dr_b2 = (const float*)d_in[26];

  // encode initial state -> z_k (ws)
  mlp(x_k, 12, 0, 3, BATCH, ep0, 32, ep1, ep2, 32, 32, ep_b0, ep_b1, ep_b2,
      zk, 128, 0, 1);
  mlp(x_k, 12, 3, 9, BATCH, er0, 32, er1, er2, 96, 96, er_b0, er_b1, er_b2,
      zk, 128, 32, 1);
  // encode target sequence -> z_target (out)
  mlp(x_next, 12, 0, 3, BM, ep0, 32, ep1, ep2, 32, 32, ep_b0, ep_b1, ep_b2,
      outp + O_ZT, 128, 0, 8);
  mlp(x_next, 12, 3, 9, BM, er0, 32, er1, er2, 96, 96, er_b0, er_b1, er_b2,
      outp + O_ZT, 128, 32, 8);
  // x_k_hat = decode(z_k)
  mlp(zk, 128, 0, 32, BATCH, dp0, 32, dp1, dp2, 16, 3, dp_b0, dp_b1, dp_b2,
      outp + O_XK, 12, 0, 1);
  mlp(zk, 128, 32, 96, BATCH, dr0, 96, dr1, dr2, 16, 9, dr_b0, dr_b1, dr_b2,
      outp + O_XK, 12, 3, 1);
  // x_target_seq_hat = decode(z_target)
  mlp(outp + O_ZT, 128, 0, 32, BM, dp0, 32, dp1, dp2, 16, 3, dp_b0, dp_b1, dp_b2,
      outp + O_XT, 12, 0, 8);
  mlp(outp + O_ZT, 128, 32, 96, BM, dr0, 96, dr1, dr2, 16, 9, dr_b0, dr_b1, dr_b2,
      outp + O_XT, 12, 3, 8);
  // Koopman recurrence -> z_pred_seq
  {
    size_t smem = (16384 + 4096 + 8192 + 8192 + 2048) * 2;
    koopman_kernel<<<dim3((unsigned)(BATCH / RPB)), TPB, smem, stream>>>(
        zk, u_seq, atf, btf, outp + O_ZP, MSTEP);
  }
  // x_pred_seq_hat = decode(z_pred)
  mlp(outp + O_ZP, 128, 0, 32, BM, dp0, 32, dp1, dp2, 16, 3, dp_b0, dp_b1, dp_b2,
      outp + O_XP, 12, 0, 8);
  mlp(outp + O_ZP, 128, 32, 96, BM, dr0, 96, dr1, dr2, 16, 9, dr_b0, dr_b1, dr_b2,
      outp + O_XP, 12, 3, 8);
}
